// GroupedQueryAttention_5265629905296
// MI455X (gfx1250) — matrix operand order repaired and checked by reading, not compiled, not run on hardware
//
#include <hip/hip_runtime.h>
#include <hip/hip_bf16.h>

// ---------------------------------------------------------------------------
// GQA forward for MI455X (gfx1250, wave32, WMMA 16x16x32 f16, f32 accum)
// ---------------------------------------------------------------------------

#define T_SEQ 2048
#define HID   4096
#define NH    32
#define NKV   8
#define HD_   128
#define QD    (NH * HD_)    // 4096
#define KD    (NKV * HD_)   // 1024

typedef _Float16 v8h  __attribute__((ext_vector_type(8)));
typedef _Float16 v16h __attribute__((ext_vector_type(16)));
typedef float    v8f  __attribute__((ext_vector_type(8)));

union Frag16 { v16h v; v8h h[2]; };

// A-matrix fragment, 16x32 f16 (M x K). Lane L in [0,16): row M=L, holds
// K=0..7 (VGPR0-3) and K=16..23 (VGPR4-7); lane L+16: same row, K=8..15,24..31.
__device__ __forceinline__ v16h load_frag_a(const _Float16* row0, int row_stride, int lane) {
    const int m = lane & 15, hf = lane >> 4;
    const _Float16* p = row0 + (size_t)m * row_stride + hf * 8;
    Frag16 f;
    f.h[0] = *reinterpret_cast<const v8h*>(p);
    f.h[1] = *reinterpret_cast<const v8h*>(p + 16);
    return f.v;
}

// B-matrix fragment, 32x16 f16 (K x N), stored K-contiguous per column:
// lane&15 = column n, lanes<16 hold K=0..7,16..23, lanes>=16 hold K=8..15,24..31.
__device__ __forceinline__ v16h load_frag_b(const _Float16* col0, int col_stride, int lane) {
    const int n = lane & 15, koff = (lane >> 4) * 8;
    const _Float16* p = col0 + (size_t)n * col_stride + koff;
    Frag16 f;
    f.h[0] = *reinterpret_cast<const v8h*>(p);
    f.h[1] = *reinterpret_cast<const v8h*>(p + 16);
    return f.v;
}

__device__ __forceinline__ v8f wmma_f16(v16h a, v16h b, v8f c) {
    return __builtin_amdgcn_wmma_f32_16x16x32_f16(false, a, false, b, (short)0, c, false, false);
}

// ---------------------------------------------------------------------------
// GEMM: C[M,N] = A[M,K] * B[K,N], A: TA (f32 or f16), B: f32, C: TC.
// Block tile 128x128, 256 threads = 8 waves (4 M x 2 N), wave tile 32x64.
// Register-staged double-buffered LDS: global loads for panel k+1 issue
// before the WMMA block for panel k, one barrier per K-step.
// ---------------------------------------------------------------------------
template <typename TA, typename TC>
__global__ __launch_bounds__(256) void gemm_wmma(const TA* __restrict__ A,
                                                 const float* __restrict__ B,
                                                 TC* __restrict__ C,
                                                 int M, int N, int K,
                                                 int lda, int ldb, int ldc) {
    constexpr int LDT = 40;  // padded row stride (halfs): 80B, 16B-aligned
    __shared__ _Float16 As[2][128 * LDT];   // 2 x 10 KB
    __shared__ _Float16 Bs[2][128 * LDT];   // 2 x 10 KB

    const int tid = threadIdx.x, lane = tid & 31, wave = tid >> 5;
    const int wm = wave >> 1, wn = wave & 1;
    const int m0 = blockIdx.y * 128, n0 = blockIdx.x * 128;

    TA    ra[16];
    float rb[16];

    auto gload = [&](int kt) {
        #pragma unroll
        for (int u = 0; u < 16; ++u) {          // A tile 128x32, k-contiguous
            const int i = tid + u * 256, m = i >> 5, k = i & 31;
            ra[u] = A[(size_t)(m0 + m) * lda + kt + k];
        }
        #pragma unroll
        for (int u = 0; u < 16; ++u) {          // B tile 32x128, n-contiguous
            const int i = tid + u * 256, k = i >> 7, n = i & 127;
            rb[u] = B[(size_t)(kt + k) * ldb + n0 + n];
        }
    };
    auto lstore = [&](int buf) {
        #pragma unroll
        for (int u = 0; u < 16; ++u) {
            const int i = tid + u * 256, m = i >> 5, k = i & 31;
            As[buf][m * LDT + k] = (_Float16)ra[u];
        }
        #pragma unroll
        for (int u = 0; u < 16; ++u) {          // store transposed: Bs[n][k]
            const int i = tid + u * 256, k = i >> 7, n = i & 127;
            Bs[buf][n * LDT + k] = (_Float16)rb[u];
        }
    };

    v8f acc[2][4] = {};

    gload(0);
    lstore(0);
    __syncthreads();

    int buf = 0;
    for (int kt = 0; kt < K; kt += 32) {
        const bool have_next = (kt + 32 < K);
        if (have_next) gload(kt + 32);          // issue loads before WMMA block
        if (kt + 64 < K) {                      // prefetch panel after next
            __builtin_prefetch(&A[(size_t)(m0 + (tid >> 1)) * lda + kt + 64], 0, 0);
            __builtin_prefetch(&B[(size_t)(kt + 64 + (tid >> 3)) * ldb + n0], 0, 0);
        }

        v16h af[2], bf[4];
        #pragma unroll
        for (int mi = 0; mi < 2; ++mi)
            af[mi] = load_frag_a(&As[buf][(wm * 32 + mi * 16) * LDT], LDT, lane);
        #pragma unroll
        for (int ni = 0; ni < 4; ++ni)
            bf[ni] = load_frag_b(&Bs[buf][(wn * 64 + ni * 16) * LDT], LDT, lane);
        #pragma unroll
        for (int mi = 0; mi < 2; ++mi)
            #pragma unroll
            for (int ni = 0; ni < 4; ++ni)
                acc[mi][ni] = wmma_f16(af[mi], bf[ni], acc[mi][ni]);

        if (have_next) lstore(buf ^ 1);
        __syncthreads();
        buf ^= 1;
    }

    // epilogue: C-frag layout  M = r + 8*(lane>=16), N = lane&15
    const int hf = lane >> 4, nl = lane & 15;
    #pragma unroll
    for (int mi = 0; mi < 2; ++mi)
        #pragma unroll
        for (int ni = 0; ni < 4; ++ni)
            #pragma unroll
            for (int r = 0; r < 8; ++r) {
                const int row = m0 + wm * 32 + mi * 16 + hf * 8 + r;
                const int col = n0 + wn * 64 + ni * 16 + nl;
                C[(size_t)row * ldc + col] = (TC)acc[mi][ni][r];
            }
}

// ---------------------------------------------------------------------------
// Llama-3 RoPE, in place on f16 Q [T, NH*HD] and K [T, NKV*HD].
// ---------------------------------------------------------------------------
__global__ __launch_bounds__(256) void rope_kernel(_Float16* __restrict__ Q,
                                                   _Float16* __restrict__ Kt) {
    const int gid = blockIdx.x * blockDim.x + threadIdx.x;
    const int per_t = (NH + NKV) * (HD_ / 2);  // 2560
    const int t = gid / per_t;
    if (t >= T_SEQ) return;
    const int rem = gid - t * per_t;
    const int slot = rem >> 6, i = rem & 63;

    _Float16* p = (slot < NH) ? (Q + (size_t)t * QD + slot * HD_)
                              : (Kt + (size_t)t * KD + (slot - NH) * HD_);

    // inv = theta^(-2i/HD) with llama3 low/high-freq scaling
    const float e = (float)(2 * i) * (1.0f / (float)HD_);
    const float inv = __expf(-e * 13.122363377404328f);   // ln(500000)
    const float wavelen = 6.2831853071795864f / inv;
    float f;
    if (wavelen > 8192.0f) {            // > ORIG_MAX / LO_FF
        f = inv * 0.125f;               // inv / FACTOR
    } else if (wavelen < 2048.0f) {     // < ORIG_MAX / HI_FF
        f = inv;
    } else {
        const float smooth = (8192.0f / wavelen - 1.0f) * (1.0f / 3.0f);
        f = (1.0f - smooth) * inv * 0.125f + smooth * inv;
    }
    const float ang = (float)t * f;
    float s, c;
    __sincosf(ang, &s, &c);
    const float x1 = (float)p[i], x2 = (float)p[i + 64];
    p[i]      = (_Float16)(x1 * c - x2 * s);
    p[i + 64] = (_Float16)(x2 * c + x1 * s);
}

// ---------------------------------------------------------------------------
// Causal GQA flash attention. 1 wave per (16-query tile, head).
// scores = Q*K^T (WMMA), online softmax in f32, O += P*V (WMMA).
// K/V fragments are preloaded into registers before each WMMA block so the
// loads clause together and the WMMAs issue back-to-back.
// ---------------------------------------------------------------------------
__global__ __launch_bounds__(32) void attn_kernel(const _Float16* __restrict__ Q,
                                                  const _Float16* __restrict__ Kt,
                                                  const _Float16* __restrict__ V,
                                                  _Float16* __restrict__ O) {
    const int lane = threadIdx.x;
    const int q0 = blockIdx.x * 16;
    const int h = blockIdx.y;
    const int kvh = h >> 2;                      // H/KV = 4
    const int hf = lane >> 4, nl = lane & 15;

    __shared__ _Float16 Plds[16 * 32];           // P tile (C-layout -> A-layout)
    __shared__ _Float16 Vlds[128 * 40];          // V^T tile [d][key], padded

    v16h qf[4];
    #pragma unroll
    for (int d = 0; d < 4; ++d)
        qf[d] = load_frag_a(Q + (size_t)q0 * QD + h * HD_ + d * 32, QD, lane);

    v8f acc[8] = {};
    float mrun[8], lrun[8];
    #pragma unroll
    for (int r = 0; r < 8; ++r) { mrun[r] = -3.0e38f; lrun[r] = 0.0f; }

    const float scale = 0.08838834764831845f;    // 1/sqrt(128)

    for (int kb = 0; kb < q0 + 16; kb += 32) {
        // preload all K fragments of this 32-key step (clause of b128 loads)
        v16h kf[2][4];
        #pragma unroll
        for (int j = 0; j < 2; ++j)
            #pragma unroll
            for (int d = 0; d < 4; ++d)
                kf[j][d] = load_frag_b(Kt + (size_t)(kb + j * 16) * KD + kvh * HD_ + d * 32,
                                       KD, lane);

        // stage V^T: Vlds[d][s] = V[kb+s][kvh*HD + d]  (overlaps score WMMAs)
        for (int i = lane; i < 32 * 128; i += 32) {
            const int s = i >> 7, d = i & 127;
            Vlds[d * 40 + s] = V[(size_t)(kb + s) * KD + kvh * HD_ + d];
        }

        // scores: two 16-key subtiles, 4 K-chunks of 32 over HD=128
        v8f sc[2] = {};
        #pragma unroll
        for (int j = 0; j < 2; ++j)
            #pragma unroll
            for (int d = 0; d < 4; ++d)
                sc[j] = wmma_f16(qf[d], kf[j][d], sc[j]);

        // online softmax (rows m = hf*8 + r, key col = nl)
        float alpha[8];
        #pragma unroll
        for (int r = 0; r < 8; ++r) {
            const int qg = q0 + hf * 8 + r;
            float s0 = sc[0][r] * scale;
            float s1 = sc[1][r] * scale;
            if (kb + nl > qg)      s0 = -3.0e38f;
            if (kb + 16 + nl > qg) s1 = -3.0e38f;
            float vmax = fmaxf(s0, s1);
            #pragma unroll
            for (int m = 1; m <= 8; m <<= 1) vmax = fmaxf(vmax, __shfl_xor(vmax, m, 32));
            const float mnew = fmaxf(mrun[r], vmax);
            const float p0 = __expf(s0 - mnew);
            const float p1 = __expf(s1 - mnew);
            float rs = p0 + p1;
            #pragma unroll
            for (int m = 1; m <= 8; m <<= 1) rs += __shfl_xor(rs, m, 32);
            alpha[r] = __expf(mrun[r] - mnew);
            lrun[r] = lrun[r] * alpha[r] + rs;
            mrun[r] = mnew;
            const int mrow = hf * 8 + r;
            Plds[mrow * 32 + nl]      = (_Float16)p0;
            Plds[mrow * 32 + 16 + nl] = (_Float16)p1;
        }
        #pragma unroll
        for (int dc = 0; dc < 8; ++dc)
            #pragma unroll
            for (int r = 0; r < 8; ++r) acc[dc][r] *= alpha[r];

        __syncthreads();  // order Plds/Vlds stores vs fragment loads

        const v16h pf = load_frag_a(Plds, 32, lane);   // P: 16x32 f16 A-frag
        v16h vf[8];
        #pragma unroll
        for (int dc = 0; dc < 8; ++dc)
            vf[dc] = load_frag_b(&Vlds[dc * 16 * 40], 40, lane);  // 32 keys x 16 dims
        #pragma unroll
        for (int dc = 0; dc < 8; ++dc)
            acc[dc] = wmma_f16(pf, vf[dc], acc[dc]);

        __syncthreads();
    }

    // epilogue: O[t, h*HD + d] layout (== transpose+reshape in reference)
    #pragma unroll
    for (int dc = 0; dc < 8; ++dc)
        #pragma unroll
        for (int r = 0; r < 8; ++r) {
            const int qg = q0 + hf * 8 + r;
            O[(size_t)qg * QD + h * HD_ + dc * 16 + nl] = (_Float16)(acc[dc][r] / lrun[r]);
        }
}

// ---------------------------------------------------------------------------
// Host launcher
// ---------------------------------------------------------------------------
extern "C" void kernel_launch(void* const* d_in, const int* in_sizes, int n_in,
                              void* d_out, int out_size, void* d_ws, size_t ws_size,
                              hipStream_t stream) {
    const float* x  = (const float*)d_in[0];
    const float* wq = (const float*)d_in[1];
    const float* wk = (const float*)d_in[2];
    const float* wv = (const float*)d_in[3];
    const float* wo = (const float*)d_in[4];
    float* out = (float*)d_out;

    char* ws = (char*)d_ws;
    const size_t bytesQ  = (size_t)T_SEQ * QD * sizeof(_Float16);  // 16 MiB
    const size_t bytesKV = (size_t)T_SEQ * KD * sizeof(_Float16);  //  4 MiB
    _Float16* Qh = (_Float16*)(ws);
    _Float16* Kh = (_Float16*)(ws + bytesQ);
    _Float16* Vh = (_Float16*)(ws + bytesQ + bytesKV);
    _Float16* Ah = (_Float16*)(ws + bytesQ + 2 * bytesKV);         // attn out, 16 MiB

    // QKV projections (f32 in -> f16 out, f32 accumulate in WMMA)
    gemm_wmma<float, _Float16><<<dim3(QD / 128, T_SEQ / 128), 256, 0, stream>>>(
        x, wq, Qh, T_SEQ, QD, HID, HID, QD, QD);
    gemm_wmma<float, _Float16><<<dim3(KD / 128, T_SEQ / 128), 256, 0, stream>>>(
        x, wk, Kh, T_SEQ, KD, HID, HID, KD, KD);
    gemm_wmma<float, _Float16><<<dim3(KD / 128, T_SEQ / 128), 256, 0, stream>>>(
        x, wv, Vh, T_SEQ, KD, HID, HID, KD, KD);

    // RoPE on Q and K in place
    {
        const int total = T_SEQ * (NH + NKV) * (HD_ / 2);
        rope_kernel<<<(total + 255) / 256, 256, 0, stream>>>(Qh, Kh);
    }

    // causal GQA attention
    attn_kernel<<<dim3(T_SEQ / 16, NH), 32, 0, stream>>>(Qh, Kh, Vh, Ah);

    // output projection (f16 in -> f32 out)
    gemm_wmma<_Float16, float><<<dim3(HID / 128, T_SEQ / 128), 256, 0, stream>>>(
        Ah, wo, out, T_SEQ, HID, QD, QD, HID, HID);
}
